// NCM_1846835937566
// MI455X (gfx1250) — matrix operand order repaired and checked by reading, not compiled, not run on hardware
//
#include <hip/hip_runtime.h>

typedef _Float16 h8  __attribute__((ext_vector_type(8)));
typedef _Float16 h16 __attribute__((ext_vector_type(16)));
typedef float    f8  __attribute__((ext_vector_type(8)));

#define N_NODES 32
#define ENDO    16
#define EXO     16
#define HID     64
#define N_ROOTS 4
#define FEAT    (N_NODES * EXO)   // 512

#define BT   128     // batch rows per workgroup
#define NTHR 256     // 8 wave32s
#define XS   104     // X LDS stride (f16); 208B = 13*16B
#define HS   72      // H LDS stride; 144B
#define WS1  104     // W1^T stride
#define WS23 72      // W2^T / W3^T stride
#define VS   24      // V ring stride (48B)

// ---- per-node weight slab in d_ws, byte layout == LDS image ----
#define OFF_W1T  0          // f16[64*104]  -> 13312 B
#define OFF_W2T  13312      // f16[64*72]   ->  9216 B
#define OFF_W3T  22528      // f16[16*72]   ->  2304 B
#define OFF_BIAS 24832      // f32[144]     ->   576 B
#define SLAB_B   25408      // multiple of 16
#define SLAB_CHUNKS (SLAB_B / 16)   // 1588 x 16B
#define U_CHUNKS    (BT * 4)        // 128 rows x 64B

__device__ __forceinline__ h16 ld_frag(const _Float16* p0, const _Float16* p1) {
    union { h16 v; h8 h[2]; } u;
    u.h[0] = *(const h8*)p0;
    u.h[1] = *(const h8*)p1;
    return u.v;
}

// CDNA5 async global->LDS copy, 16B per lane, tracked by ASYNCcnt.
__device__ __forceinline__ void async_copy16(unsigned lds_addr, unsigned long long gbase, unsigned goff) {
    asm volatile("global_load_async_to_lds_b128 %0, %1, %2"
                 :: "v"(lds_addr), "v"(goff), "s"(gbase) : "memory");
}
__device__ __forceinline__ void wait_async0() {
    asm volatile("s_wait_asynccnt 0x0" ::: "memory");
}

__device__ __forceinline__ void prefetch_slab(const unsigned char* gslab, const unsigned char* lbase, int tid) {
    unsigned lds0 = (unsigned)(size_t)lbase;
    unsigned long long gb = (unsigned long long)(size_t)gslab;
    for (int c = tid; c < SLAB_CHUNKS; c += NTHR)
        async_copy16(lds0 + (unsigned)c * 16u, gb, (unsigned)c * 16u);
}

__device__ __forceinline__ void prefetch_u(const float* u, const float* lbase, int tid, int node, int rowbase) {
    unsigned lds0 = (unsigned)(size_t)lbase;
    unsigned long long gb = (unsigned long long)(size_t)(u + (size_t)rowbase * FEAT + node * EXO);
    for (int c = tid; c < U_CHUNKS; c += NTHR) {
        int r = c >> 2, q = c & 3;
        async_copy16(lds0 + (unsigned)c * 16u, gb, (unsigned)(r * FEAT * 4 + q * 16));
    }
}

// ---------------- prep kernel: f32 weights -> f16 transposed/padded slabs ----------------
__global__ __launch_bounds__(NTHR)
void pack_weights(const float* __restrict__ wr1, const float* __restrict__ br1,
                  const float* __restrict__ wr2, const float* __restrict__ br2,
                  const float* __restrict__ wr3, const float* __restrict__ br3,
                  const float* __restrict__ wi1, const float* __restrict__ bi1,
                  const float* __restrict__ wi2, const float* __restrict__ bi2,
                  const float* __restrict__ wi3, const float* __restrict__ bi3,
                  unsigned char* __restrict__ ws)
{
    const int node = blockIdx.x;
    const int tid  = threadIdx.x;
    const bool root = node < N_ROOTS;
    const int  j    = node - N_ROOTS;
    const int  k1   = root ? EXO : 80;

    const float* w1 = root ? (wr1 + node * (EXO * HID))  : (wi1 + (size_t)j * (80 * HID));
    const float* w2 = root ? (wr2 + node * (HID * HID))  : (wi2 + (size_t)j * (HID * HID));
    const float* w3 = root ? (wr3 + node * (HID * ENDO)) : (wi3 + (size_t)j * (HID * ENDO));
    const float* b1 = root ? (br1 + node * HID)  : (bi1 + j * HID);
    const float* b2 = root ? (br2 + node * HID)  : (bi2 + j * HID);
    const float* b3 = root ? (br3 + node * ENDO) : (bi3 + j * ENDO);

    unsigned char* slab = ws + (size_t)node * SLAB_B;
    _Float16* W1T = (_Float16*)(slab + OFF_W1T);
    _Float16* W2T = (_Float16*)(slab + OFF_W2T);
    _Float16* W3T = (_Float16*)(slab + OFF_W3T);
    float*    Bss = (float*)   (slab + OFF_BIAS);

    for (int idx = tid; idx < HID * WS1; idx += NTHR) {
        int n = idx / WS1, k = idx % WS1;
        W1T[idx] = (k < k1) ? (_Float16)w1[k * HID + n] : (_Float16)0.f;
    }
    for (int idx = tid; idx < HID * WS23; idx += NTHR) {
        int n = idx / WS23, k = idx % WS23;
        W2T[idx] = (k < HID) ? (_Float16)w2[k * HID + n] : (_Float16)0.f;
    }
    for (int idx = tid; idx < ENDO * WS23; idx += NTHR) {
        int n = idx / WS23, k = idx % WS23;
        W3T[idx] = (k < HID) ? (_Float16)w3[k * ENDO + n] : (_Float16)0.f;
    }
    if (tid < HID)                 Bss[tid] = b1[tid];
    else if (tid < 2 * HID)        Bss[tid] = b2[tid - HID];
    else if (tid < 2 * HID + ENDO) Bss[tid] = b3[tid - 2 * HID];
}

// ---------------- fused per-node 3-layer MLP (per-wave 16-row strip) ----------------
template<int NK1>
__device__ __forceinline__ void compute_node(const _Float16* Xs, _Float16* Hsm,
                                             const _Float16* W1T, const _Float16* W2T,
                                             const _Float16* W3T, const float* Bss,
                                             float* outp, _Float16* vring,
                                             int strip, int nl, int hg)
{
    const int arow = strip + nl;

    // Layer 1: X(16 x 32*NK1) @ W1 -> ReLU -> Hsm
    {
        f8 acc[4];
        #pragma unroll
        for (int t = 0; t < 4; ++t) {
            float bv = Bss[t * 16 + nl];
            #pragma unroll
            for (int v = 0; v < 8; ++v) acc[t][v] = bv;
        }
        #pragma unroll
        for (int s = 0; s < NK1; ++s) {
            const int kk = s * 32;
            h16 a = ld_frag(&Xs[arow * XS + kk + hg * 8],
                            &Xs[arow * XS + kk + hg * 8 + 16]);
            #pragma unroll
            for (int t = 0; t < 4; ++t) {
                h16 b = ld_frag(&W1T[(t * 16 + nl) * WS1 + kk + hg * 8],
                                &W1T[(t * 16 + nl) * WS1 + kk + hg * 8 + 16]);
                acc[t] = __builtin_amdgcn_wmma_f32_16x16x32_f16(
                    false, a, false, b, (short)0, acc[t], false, false);
            }
        }
        #pragma unroll
        for (int t = 0; t < 4; ++t)
            #pragma unroll
            for (int v = 0; v < 8; ++v)
                Hsm[(strip + v + 8 * hg) * HS + t * 16 + nl] = (_Float16)fmaxf(acc[t][v], 0.f);
    }

    // Layer 2: H(16x64) @ W2(64x64) -> ReLU -> Hsm (same-wave DS ops stay in order)
    {
        f8 acc[4];
        #pragma unroll
        for (int t = 0; t < 4; ++t) {
            float bv = Bss[HID + t * 16 + nl];
            #pragma unroll
            for (int v = 0; v < 8; ++v) acc[t][v] = bv;
        }
        #pragma unroll
        for (int s = 0; s < 2; ++s) {
            const int kk = s * 32;
            h16 a = ld_frag(&Hsm[arow * HS + kk + hg * 8],
                            &Hsm[arow * HS + kk + hg * 8 + 16]);
            #pragma unroll
            for (int t = 0; t < 4; ++t) {
                h16 b = ld_frag(&W2T[(t * 16 + nl) * WS23 + kk + hg * 8],
                                &W2T[(t * 16 + nl) * WS23 + kk + hg * 8 + 16]);
                acc[t] = __builtin_amdgcn_wmma_f32_16x16x32_f16(
                    false, a, false, b, (short)0, acc[t], false, false);
            }
        }
        #pragma unroll
        for (int t = 0; t < 4; ++t)
            #pragma unroll
            for (int v = 0; v < 8; ++v)
                Hsm[(strip + v + 8 * hg) * HS + t * 16 + nl] = (_Float16)fmaxf(acc[t][v], 0.f);
    }

    // Layer 3: H(16x64) @ W3(64x16); logits -> HBM, thresholded bits -> ring
    {
        f8 acc;
        {
            float bv = Bss[2 * HID + nl];
            #pragma unroll
            for (int v = 0; v < 8; ++v) acc[v] = bv;
        }
        #pragma unroll
        for (int s = 0; s < 2; ++s) {
            const int kk = s * 32;
            h16 a = ld_frag(&Hsm[arow * HS + kk + hg * 8],
                            &Hsm[arow * HS + kk + hg * 8 + 16]);
            h16 b = ld_frag(&W3T[nl * WS23 + kk + hg * 8],
                            &W3T[nl * WS23 + kk + hg * 8 + 16]);
            acc = __builtin_amdgcn_wmma_f32_16x16x32_f16(
                false, a, false, b, (short)0, acc, false, false);
        }
        #pragma unroll
        for (int v = 0; v < 8; ++v) {
            int m = v + 8 * hg;
            float val = acc[v];
            outp[(size_t)(strip + m) * FEAT + nl] = val;
            vring[(strip + m) * VS + nl] = (_Float16)(val > 0.5f ? 1.f : 0.f);
        }
    }
}

__global__ __launch_bounds__(NTHR)
void ncm_chain_kernel(const float* __restrict__ u,
                      const unsigned char* __restrict__ ws,
                      float* __restrict__ out)
{
    __shared__ __align__(16) _Float16 Xs [BT * XS];
    __shared__ __align__(16) _Float16 Hsm[BT * HS];
    __shared__ __align__(16) unsigned char Wb[2][SLAB_B];   // double-buffered weight slabs
    __shared__ __align__(16) float Ub[2][BT * EXO];         // double-buffered exogenous block (f32)
    __shared__ __align__(16) _Float16 Vr[4][BT * VS];       // thresholded parent ring

    const int tid   = threadIdx.x;
    const int lane  = tid & 31;
    const int wave  = tid >> 5;
    const int nl    = lane & 15;
    const int hg    = lane >> 4;
    const int strip = wave * 16;
    const int rowbase = blockIdx.x * BT;

    // prologue: zero X once (pad cols 80..96 / root cols 16..32 stay zero forever),
    // and pull node-0 slab + u-block via the async pipe.
    prefetch_slab(ws, &Wb[0][0], tid);
    prefetch_u(u, &Ub[0][0], tid, 0, rowbase);
    {
        h8 z = {};
        for (int c8 = tid; c8 < (BT * XS) / 8; c8 += NTHR)
            *((h8*)Xs + c8) = z;
    }
    wait_async0();
    __syncthreads();

    for (int node = 0; node < N_NODES; ++node) {
        const int cur = node & 1, nxt = cur ^ 1;
        const bool root = node < N_ROOTS;

        // stream node+1 slab + u-block into the other buffers while this node runs
        if (node + 1 < N_NODES) {
            prefetch_slab(ws + (size_t)(node + 1) * SLAB_B, &Wb[nxt][0], tid);
            prefetch_u(u, &Ub[nxt][0], tid, node + 1, rowbase);
        }

        // ---- per-wave X build: each wave assembles only its own 16-row strip ----
        // (ring rows were written by this same wave last iteration -> in-order DS,
        //  Ub was fenced by the previous wait_async + barrier; no mid-node barrier needed)
        const int xrow = strip + (lane & 15);
        const int half = lane >> 4;
        if (!root) {
            #pragma unroll
            for (int k = 0; k < 4; ++k) {
                int unit = lane + 32 * k;          // 0..127
                int row  = strip + (unit & 15);
                int rest = unit >> 4;              // 0..7
                int p    = rest & 3;
                int ph   = rest >> 2;
                int slot = (node - 4 + p) & 3;
                *(h8*)&Xs[row * XS + p * 16 + ph * 8] =
                    *(const h8*)&Vr[slot][row * VS + ph * 8];
            }
        }
        {
            const int xcol = (root ? 0 : 64) + half * 8;
            const float* usrc = &Ub[cur][xrow * EXO + half * 8];
            h8 hv;
            #pragma unroll
            for (int e = 0; e < 8; ++e) hv[e] = (_Float16)usrc[e];
            *(h8*)&Xs[xrow * XS + xcol] = hv;
        }

        const _Float16* W1T = (const _Float16*)(&Wb[cur][OFF_W1T]);
        const _Float16* W2T = (const _Float16*)(&Wb[cur][OFF_W2T]);
        const _Float16* W3T = (const _Float16*)(&Wb[cur][OFF_W3T]);
        const float*    Bss = (const float*)   (&Wb[cur][OFF_BIAS]);
        float* outp = out + (size_t)rowbase * FEAT + node * ENDO;
        _Float16* vring = &Vr[node & 3][0];

        if (root)
            compute_node<1>(Xs, Hsm, W1T, W2T, W3T, Bss, outp, vring, strip, nl, hg);
        else
            compute_node<3>(Xs, Hsm, W1T, W2T, W3T, Bss, outp, vring, strip, nl, hg);

        wait_async0();     // node+1 slab/u landed in LDS
        __syncthreads();   // next-node buffers visible to all waves
    }
}

extern "C" void kernel_launch(void* const* d_in, const int* in_sizes, int n_in,
                              void* d_out, int out_size, void* d_ws, size_t ws_size,
                              hipStream_t stream) {
    const float* u   = (const float*)d_in[0];
    const float* wr1 = (const float*)d_in[1];
    const float* br1 = (const float*)d_in[2];
    const float* wr2 = (const float*)d_in[3];
    const float* br2 = (const float*)d_in[4];
    const float* wr3 = (const float*)d_in[5];
    const float* br3 = (const float*)d_in[6];
    const float* wi1 = (const float*)d_in[7];
    const float* bi1 = (const float*)d_in[8];
    const float* wi2 = (const float*)d_in[9];
    const float* bi2 = (const float*)d_in[10];
    const float* wi3 = (const float*)d_in[11];
    const float* bi3 = (const float*)d_in[12];
    float* out = (float*)d_out;
    unsigned char* ws = (unsigned char*)d_ws;

    const int B = in_sizes[0] / FEAT;

    hipLaunchKernelGGL(pack_weights, dim3(N_NODES), dim3(NTHR), 0, stream,
                       wr1, br1, wr2, br2, wr3, br3,
                       wi1, bi1, wi2, bi2, wi3, bi3, ws);
    hipLaunchKernelGGL(ncm_chain_kernel, dim3(B / BT), dim3(NTHR), 0, stream,
                       u, ws, out);
}
